// LSTMBase_88115549045329
// MI455X (gfx1250) — matrix operand order repaired and checked by reading, not compiled, not run on hardware
//
#include <hip/hip_runtime.h>
#include <hip/hip_bf16.h>

// ---------------------------------------------------------------------------
// 2-layer LSTM on gfx1250 (MI455X), bf16 WMMA with f32 accumulate.
//   T=512, B=64, I=H=1024, gates [i,f,g,o] along 4H.
// ---------------------------------------------------------------------------

typedef __attribute__((ext_vector_type(16))) __bf16 v16bf;
typedef __attribute__((ext_vector_type(8)))  __bf16 v8bf;
typedef __attribute__((ext_vector_type(4)))  __bf16 v4bf;
typedef __attribute__((ext_vector_type(8)))  float  v8f;
typedef __attribute__((ext_vector_type(4)))  float  v4f;

#define LSTM_T 512
#define LSTM_B 64
#define LSTM_H 1024
#define LSTM_NG 4096  // 4*H

union BF16x16 {
    v16bf v;
    struct { v8bf lo; v8bf hi; } p;
};

// A fragment (16x32 bf16, M = lane%16): per-lane K ranges
//   V0..3 : k0 + half*16 .. +7     V4..7 : k0 + half*16 + 8 .. +7
__device__ __forceinline__ v16bf load_a_frag(const __bf16* __restrict__ rowp,
                                             int k0, int half) {
    BF16x16 u;
    u.p.lo = *(const v8bf*)(rowp + k0 + half * 16);
    u.p.hi = *(const v8bf*)(rowp + k0 + half * 16 + 8);
    return u.v;
}

// B fragment (32x16 bf16, N = lane%16): per-lane K range k0 + half*16 .. +15
// (contiguous 32 bytes from row-major W[n][k])
__device__ __forceinline__ v16bf load_b_frag(const __bf16* __restrict__ rowp,
                                             int k0, int half) {
    return *(const v16bf*)(rowp + k0 + half * 16);
}

__device__ __forceinline__ float sigm_f(float x) {
    return 1.0f / (1.0f + __expf(-x));
}
__device__ __forceinline__ float tanh_f(float x) {
    // 1 - 2/(e^{2x}+1): saturates cleanly to +-1 at extremes
    return 1.0f - 2.0f / (__expf(2.0f * x) + 1.0f);
}

#define WMMA_BF16(a, b, c) \
    __builtin_amdgcn_wmma_f32_16x16x32_bf16(false, (a), false, (b), (short)0, (c), false, false)

// ---------------------------------------------------------------------------
// Big GEMM: xg[M][4096] (bf16) = A[M][1024](bf16) @ W[4096][1024]^T + bias
// 2x2 register tiling per wave: workgroup tile = 32 M x 128 N,
// wave w covers N = [n0 + w*32, +32).  Grid: (4096/128, M/32), block 128.
// ---------------------------------------------------------------------------
__global__ __launch_bounds__(128)
void gemm_xg_kernel(__bf16* __restrict__ xg, const __bf16* __restrict__ A,
                    const __bf16* __restrict__ W, const float* __restrict__ bias) {
    const int tid  = threadIdx.x;
    const int wave = tid >> 5;
    const int lane = tid & 31;
    const int lm   = lane & 15;
    const int half = lane >> 4;
    const int K    = LSTM_H;

    const int n0 = blockIdx.x * 128 + wave * 32;
    const int m0 = blockIdx.y * 32;

    const float bv0 = bias[n0 + lm];
    const float bv1 = bias[n0 + 16 + lm];
    v8f acc00, acc01, acc10, acc11;
#pragma unroll
    for (int r = 0; r < 8; ++r) {
        acc00[r] = bv0; acc01[r] = bv1;
        acc10[r] = bv0; acc11[r] = bv1;
    }

    const __bf16* arow0 = A + (size_t)(m0 + lm) * K;
    const __bf16* arow1 = arow0 + (size_t)16 * K;
    const __bf16* brow0 = W + (size_t)(n0 + lm) * K;
    const __bf16* brow1 = brow0 + (size_t)16 * K;

#pragma unroll 2
    for (int k0 = 0; k0 < K; k0 += 32) {
        v16bf a0 = load_a_frag(arow0, k0, half);
        v16bf a1 = load_a_frag(arow1, k0, half);
        v16bf b0 = load_b_frag(brow0, k0, half);
        v16bf b1 = load_b_frag(brow1, k0, half);
        acc00 = WMMA_BF16(a0, b0, acc00);
        acc01 = WMMA_BF16(a0, b1, acc01);
        acc10 = WMMA_BF16(a1, b0, acc10);
        acc11 = WMMA_BF16(a1, b1, acc11);
    }

    // D layout: element (Mrow,N) -> VGPR Mrow%8, lane (Mrow/8)*16 + N.
    // xg is streamed (read once later): nontemporal stores.
    v8f* accs[4] = {&acc00, &acc01, &acc10, &acc11};
#pragma unroll
    for (int mi = 0; mi < 2; ++mi) {
#pragma unroll
        for (int ni = 0; ni < 2; ++ni) {
            v8f a = *accs[mi * 2 + ni];
            __bf16* obase = xg + (size_t)(m0 + mi * 16 + half * 8) * LSTM_NG +
                            n0 + ni * 16 + lm;
#pragma unroll
            for (int r = 0; r < 8; ++r)
                __builtin_nontemporal_store((__bf16)a[r], obase + (size_t)r * LSTM_NG);
        }
    }
}

// ---------------------------------------------------------------------------
// One LSTM timestep: gates = xg[t] + h_prev @ W_hh^T, fused cell update.
// Grid: (H/16, B/32), block 128. Wave w = gate w; each wave computes TWO
// 16x16 M-tiles sharing one B fragment (halves W_hh L2 traffic).
// All 4 waves publish to LDS; all 128 threads split the cell update
// (4 consecutive columns per thread -> vectorized c/h/out accesses).
// ---------------------------------------------------------------------------
__global__ __launch_bounds__(128)
void lstm_step_kernel(const __bf16* __restrict__ hprev,  // [B][H] bf16
                      const __bf16* __restrict__ Whh,    // [4H][H] bf16
                      const __bf16* __restrict__ xg,     // [B][4H] bf16 (this t)
                      float*        __restrict__ cbuf,   // [B][H] f32 (in/out)
                      __bf16*       __restrict__ hnext,  // [B][H] bf16
                      float*        __restrict__ outf) { // [B][H] f32 or null
    __shared__ float lds[4 * 512];  // [gate][mtile 0/1][16][16]

    const int tid  = threadIdx.x;
    const int wave = tid >> 5;
    const int lane = tid & 31;
    const int lm   = lane & 15;
    const int half = lane >> 4;

    const int n0 = blockIdx.x * 16;            // h-unit tile
    const int m0 = blockIdx.y * 32;            // batch tile pair
    const int ncol = wave * LSTM_H + n0 + lm;  // gate column

    // init accumulators from precomputed x-projection (+bias); xg is
    // single-use streamed data -> nontemporal.
    v8f acc0, acc1;
    const __bf16* xg0 = xg + (size_t)(m0 + half * 8) * LSTM_NG + ncol;
    const __bf16* xg1 = xg + (size_t)(m0 + 16 + half * 8) * LSTM_NG + ncol;
#pragma unroll
    for (int r = 0; r < 8; ++r) {
        acc0[r] = (float)__builtin_nontemporal_load(xg0 + (size_t)r * LSTM_NG);
        acc1[r] = (float)__builtin_nontemporal_load(xg1 + (size_t)r * LSTM_NG);
    }

    const __bf16* arow0 = hprev + (size_t)(m0 + lm) * LSTM_H;
    const __bf16* arow1 = arow0 + (size_t)16 * LSTM_H;
    const __bf16* brow  = Whh + (size_t)ncol * LSTM_H;  // row n of W_hh (K-major)

#pragma unroll 4
    for (int k0 = 0; k0 < LSTM_H; k0 += 32) {
        v16bf b  = load_b_frag(brow, k0, half);   // shared by both M-tiles
        v16bf a0 = load_a_frag(arow0, k0, half);
        v16bf a1 = load_a_frag(arow1, k0, half);
        acc0 = WMMA_BF16(a0, b, acc0);
        acc1 = WMMA_BF16(a1, b, acc1);
    }

    // publish gate tiles
    float* tl = &lds[wave * 512];
#pragma unroll
    for (int r = 0; r < 8; ++r) {
        tl[(half * 8 + r) * 16 + lm]       = acc0[r];
        tl[256 + (half * 8 + r) * 16 + lm] = acc1[r];
    }
    __syncthreads();

    // fused LSTM cell update: 512 elements / 128 threads = 4 consecutive cols
    const int e0  = tid * 4;
    const int mt  = e0 >> 8;
    const int rem = e0 & 255;
    const int Ml  = rem >> 4;
    const int N   = rem & 15;
    const size_t idx = (size_t)(m0 + mt * 16 + Ml) * LSTM_H + n0 + N;

    v4f cv = *(const v4f*)&cbuf[idx];
    v4f cn, hn;
#pragma unroll
    for (int j = 0; j < 4; ++j) {
        float iv = sigm_f(lds[0 * 512 + e0 + j]);
        float fv = sigm_f(lds[1 * 512 + e0 + j]);
        float gv = tanh_f(lds[2 * 512 + e0 + j]);
        float ov = sigm_f(lds[3 * 512 + e0 + j]);
        cn[j] = fv * cv[j] + iv * gv;
        hn[j] = ov * tanh_f(cn[j]);
    }
    *(v4f*)&cbuf[idx] = cn;

    v4bf hb;
#pragma unroll
    for (int j = 0; j < 4; ++j) hb[j] = (__bf16)hn[j];
    *(v4bf*)&hnext[idx] = hb;

    if (outf) __builtin_nontemporal_store(hn, (v4f*)&outf[idx]);
}

// ---------------------------------------------------------------------------
// Utility kernels
// ---------------------------------------------------------------------------
__global__ void cvt_f32_to_bf16_kernel(__bf16* __restrict__ dst,
                                       const float* __restrict__ src, int n) {
    for (int i = blockIdx.x * blockDim.x + threadIdx.x; i < n;
         i += gridDim.x * blockDim.x)
        dst[i] = (__bf16)src[i];
}

__global__ void cvt_bf16_to_f32_kernel(float* __restrict__ dst,
                                       const __bf16* __restrict__ src, int n) {
    for (int i = blockIdx.x * blockDim.x + threadIdx.x; i < n;
         i += gridDim.x * blockDim.x)
        dst[i] = (float)src[i];
}

__global__ void add_f32_kernel(float* __restrict__ dst, const float* __restrict__ a,
                               const float* __restrict__ b, int n) {
    for (int i = blockIdx.x * blockDim.x + threadIdx.x; i < n;
         i += gridDim.x * blockDim.x)
        dst[i] = a[i] + b[i];
}

__global__ void copy_f32_kernel(float* __restrict__ dst,
                                const float* __restrict__ src, int n) {
    for (int i = blockIdx.x * blockDim.x + threadIdx.x; i < n;
         i += gridDim.x * blockDim.x)
        dst[i] = src[i];
}

// ---------------------------------------------------------------------------
extern "C" void kernel_launch(void* const* d_in, const int* in_sizes, int n_in,
                              void* d_out, int out_size, void* d_ws, size_t ws_size,
                              hipStream_t stream) {
    (void)in_sizes; (void)n_in; (void)out_size; (void)ws_size;

    const int T = LSTM_T, B = LSTM_B, H = LSTM_H, NG = LSTM_NG;
    const int BH = B * H;           // 65536
    const size_t M = (size_t)T * B; // 32768 rows for the big GEMMs

    const float* x      = (const float*)d_in[0];
    const float* h0     = (const float*)d_in[1];  // [2,B,H]
    const float* c0     = (const float*)d_in[2];  // [2,B,H]
    const float* W_ih_0 = (const float*)d_in[3];
    const float* W_hh_0 = (const float*)d_in[4];
    const float* b_ih_0 = (const float*)d_in[5];
    const float* b_hh_0 = (const float*)d_in[6];
    const float* W_ih_1 = (const float*)d_in[7];
    const float* W_hh_1 = (const float*)d_in[8];
    const float* b_ih_1 = (const float*)d_in[9];
    const float* b_hh_1 = (const float*)d_in[10];
    float* out = (float*)d_out;  // [T,B,H] out1 | [2,B,H] h_n | [2,B,H] c_n

    // ---- workspace carve-up --------------------------------------------
    char* base = (char*)d_ws;
    size_t off = 0;
    auto take = [&](size_t bytes) -> char* {
        char* p = base + off;
        off = (off + bytes + 255) & ~(size_t)255;
        return p;
    };
    const size_t wbytes = (size_t)NG * H * sizeof(__bf16);  // 8 MB each
    __bf16* Wb0ih = (__bf16*)take(wbytes);
    __bf16* Wb0hh = (__bf16*)take(wbytes);
    __bf16* Wb1ih = (__bf16*)take(wbytes);
    __bf16* Wb1hh = (__bf16*)take(wbytes);
    __bf16* xb    = (__bf16*)take(M * H * sizeof(__bf16));    // 64 MB
    __bf16* xg    = (__bf16*)take(M * NG * sizeof(__bf16));   // 256 MB
    __bf16* out0b = (__bf16*)take(M * H * sizeof(__bf16));    // 64 MB
    __bf16* hinit0 = (__bf16*)take((size_t)BH * sizeof(__bf16));
    __bf16* hinit1 = (__bf16*)take((size_t)BH * sizeof(__bf16));
    __bf16* hA     = (__bf16*)take((size_t)BH * sizeof(__bf16));
    __bf16* hB     = (__bf16*)take((size_t)BH * sizeof(__bf16));
    float*  c0buf  = (float*)take((size_t)BH * sizeof(float));
    float*  c1buf  = (float*)take((size_t)BH * sizeof(float));
    float*  bias0  = (float*)take((size_t)NG * sizeof(float));
    float*  bias1  = (float*)take((size_t)NG * sizeof(float));

    const int CT = 256;  // threads for utility kernels
    auto blocks = [&](int n) { int g = (n + CT - 1) / CT; return g > 4096 ? 4096 : g; };

    // ---- precision conversion / setup ----------------------------------
    cvt_f32_to_bf16_kernel<<<blocks((int)(M * H)), CT, 0, stream>>>(xb, x, (int)(M * H));
    cvt_f32_to_bf16_kernel<<<blocks(NG * H), CT, 0, stream>>>(Wb0ih, W_ih_0, NG * H);
    cvt_f32_to_bf16_kernel<<<blocks(NG * H), CT, 0, stream>>>(Wb0hh, W_hh_0, NG * H);
    cvt_f32_to_bf16_kernel<<<blocks(NG * H), CT, 0, stream>>>(Wb1ih, W_ih_1, NG * H);
    cvt_f32_to_bf16_kernel<<<blocks(NG * H), CT, 0, stream>>>(Wb1hh, W_hh_1, NG * H);
    add_f32_kernel<<<blocks(NG), CT, 0, stream>>>(bias0, b_ih_0, b_hh_0, NG);
    add_f32_kernel<<<blocks(NG), CT, 0, stream>>>(bias1, b_ih_1, b_hh_1, NG);
    cvt_f32_to_bf16_kernel<<<blocks(BH), CT, 0, stream>>>(hinit0, h0, BH);
    cvt_f32_to_bf16_kernel<<<blocks(BH), CT, 0, stream>>>(hinit1, h0 + BH, BH);
    copy_f32_kernel<<<blocks(BH), CT, 0, stream>>>(c0buf, c0, BH);
    copy_f32_kernel<<<blocks(BH), CT, 0, stream>>>(c1buf, c0 + BH, BH);

    const dim3 gemmGrid(NG / 128, (unsigned)(M / 32));  // (32, 1024)
    const dim3 stepGrid(H / 16, B / 32);                // (64, 2)

    // ---- layer 0 --------------------------------------------------------
    gemm_xg_kernel<<<gemmGrid, 128, 0, stream>>>(xg, xb, Wb0ih, bias0);
    for (int t = 0; t < T; ++t) {
        const __bf16* hprev = (t == 0) ? hinit0 : out0b + (size_t)(t - 1) * BH;
        __bf16* hnext = out0b + (size_t)t * BH;  // h chain == layer-1 input
        lstm_step_kernel<<<stepGrid, 128, 0, stream>>>(
            hprev, Wb0hh, xg + (size_t)t * B * NG, c0buf, hnext, nullptr);
    }

    // ---- layer 1 --------------------------------------------------------
    gemm_xg_kernel<<<gemmGrid, 128, 0, stream>>>(xg, out0b, Wb1ih, bias1);
    for (int t = 0; t < T; ++t) {
        const __bf16* hprev =
            (t == 0) ? hinit1 : (((t - 1) & 1) ? hB : hA);
        __bf16* hnext = (t & 1) ? hB : hA;
        lstm_step_kernel<<<stepGrid, 128, 0, stream>>>(
            hprev, Wb1hh, xg + (size_t)t * B * NG, c1buf, hnext,
            out + (size_t)t * BH);
    }

    // ---- epilogue: h_n [2,B,H], c_n [2,B,H] -----------------------------
    float* hn_out = out + (size_t)T * BH;
    cvt_bf16_to_f32_kernel<<<blocks(BH), CT, 0, stream>>>(
        hn_out, out0b + (size_t)(T - 1) * BH, BH);            // h_n layer 0
    cvt_bf16_to_f32_kernel<<<blocks(BH), CT, 0, stream>>>(
        hn_out + BH, hB, BH);                                 // h_n layer 1 (t=511 -> hB)
    copy_f32_kernel<<<blocks(BH), CT, 0, stream>>>(hn_out + 2 * BH, c0buf, BH);
    copy_f32_kernel<<<blocks(BH), CT, 0, stream>>>(hn_out + 3 * BH, c1buf, BH);
}
